// Weight_3058016715368
// MI455X (gfx1250) — matrix operand order repaired and checked by reading, not compiled, not run on hardware
//
#include <hip/hip_runtime.h>

// ---------------------------------------------------------------------------
// MI455X (gfx1250) fused Qwen3-style gate kernel:
//   q = RMSNorm(hidden @ Wq^T + bq), k = RMSNorm(hidden @ Wk^T)
//   logit = (k . q)/sqrt(D) + 2b ; logit_base[t] = (k_base[t] . q)/sqrt(D)
//   out[h,s] = mean_g 1/(1 + sum_t exp(logit_base - logit))
// GEMMs: v_wmma_f32_16x16x32_bf16; staging: global_load_async_to_lds_b128
// double-buffered in LDS with one workgroup barrier per K-chunk.
// ---------------------------------------------------------------------------

#define NSEQ   32768
#define DIN    2048
#define NHEAD  8
#define NGROUP 4
#define DHEAD  128
#define SINK   16
#define QCOLS  (NGROUP * DHEAD)   // 512
#define COLS   (QCOLS + DHEAD)    // 640 output columns per head (q cols + k cols)
#define KC     32                 // WMMA K per step (bf16)
#define TILES  (COLS / 16)        // 40 16-wide column tiles
#define NWAVE  8
#define TPW    (TILES / NWAVE)    // 5 tiles per wave

// LDS carve (bytes). Single __shared__ array => LDS offsets start at 0.
#define A0_OFF  0                        // 16 x 32 bf16  = 1024
#define A1_OFF  1024                     // 16 x 32 bf16  = 1024
#define B0_OFF  2048                     // 640 x 32 bf16 = 40960
#define B1_OFF  43008                    // 640 x 32 bf16 = 40960 (ends 83968)
#define KB_OFF  83968                    // 16 x 128 f32  = 8192  (ends 92160)
#define QK_STRIDE 644                    // padded f32 row stride (bank spread)
#define SC_OFF  (16 * QK_STRIDE * 4)     // 41216; scores 16x4 f32 = 256 (< B1_OFF)
#define SMEM_BYTES 92160

typedef __attribute__((ext_vector_type(16))) __bf16 v16bf;
typedef __attribute__((ext_vector_type(8)))  float  v8f;

__device__ __forceinline__ unsigned short f2bf(float f) {
  unsigned int u = __float_as_uint(f);
  u += 0x7FFFu + ((u >> 16) & 1u);   // round-to-nearest-even
  return (unsigned short)(u >> 16);
}

// Async global->LDS 16B copy (CDNA5 GLOBAL_LOAD_ASYNC_TO_LDS_B128, GV mode).
__device__ __forceinline__ void async_g2l_b128(unsigned ldsoff,
                                               unsigned long long gaddr) {
  asm volatile("global_load_async_to_lds_b128 %0, %1, off"
               :
               : "v"(ldsoff), "v"(gaddr)
               : "memory");
}
__device__ __forceinline__ void wait_asynccnt0() {
  asm volatile("s_wait_asynccnt 0x0" ::: "memory");
}

// ---- pre-pass 1: hidden f32 -> bf16 ---------------------------------------
__global__ __launch_bounds__(256) void cvt_hidden_bf16(
    const float* __restrict__ src, unsigned short* __restrict__ dst) {
  long long i = (long long)blockIdx.x * 256 + threadIdx.x;   // group of 4
  float4 v = ((const float4*)src)[i];
  uint2 o;
  o.x = (unsigned)f2bf(v.x) | ((unsigned)f2bf(v.y) << 16);
  o.y = (unsigned)f2bf(v.z) | ((unsigned)f2bf(v.w) << 16);
  ((uint2*)dst)[i] = o;
}

// ---- pre-pass 2: pack Wq+Wk per head into bf16 [NHEAD][COLS][DIN] ---------
__global__ __launch_bounds__(256) void pack_weights_bf16(
    const float* __restrict__ Wq, const float* __restrict__ Wk,
    unsigned short* __restrict__ dst) {
  long long i4  = (long long)blockIdx.x * 256 + threadIdx.x;  // group of 4
  long long idx = i4 * 4;
  int h = (int)(idx / ((long long)COLS * DIN));
  long long rem = idx - (long long)h * COLS * DIN;
  int n = (int)(rem / DIN);
  int k = (int)(rem % DIN);
  const float* src = (n < QCOLS)
      ? (Wq + ((size_t)(h * QCOLS + n)) * DIN + k)
      : (Wk + ((size_t)(h * DHEAD + (n - QCOLS))) * DIN + k);
  float4 v = *(const float4*)src;
  uint2 o;
  o.x = (unsigned)f2bf(v.x) | ((unsigned)f2bf(v.y) << 16);
  o.y = (unsigned)f2bf(v.z) | ((unsigned)f2bf(v.w) << 16);
  ((uint2*)dst)[i4] = o;
}

// ---- fused GEMM + RMSNorm + sink-gate kernel ------------------------------
__global__ __launch_bounds__(256) void fused_qk_gate(
    const unsigned short* __restrict__ hb,   // bf16 hidden [NSEQ][DIN]
    const unsigned short* __restrict__ wc,   // bf16 packed [NHEAD][COLS][DIN]
    const float* __restrict__ bq,            // [NHEAD*QCOLS]
    const float* __restrict__ qw,            // [DHEAD]
    const float* __restrict__ kw,            // [DHEAD]
    const float* __restrict__ bb,            // [NHEAD][1][NGROUP]
    const float* __restrict__ kbase,         // [NHEAD][1][SINK][DHEAD]
    float* __restrict__ out) {               // [NHEAD][NSEQ]
  __shared__ __align__(16) unsigned char smem[SMEM_BYTES];

  const int t    = threadIdx.x;
  const int lane = t & 31;
  const int wave = t >> 5;
  const int s0   = blockIdx.x * 16;
  const int h    = blockIdx.y;

  // stage this head's k_base (16x128 f32) into LDS
  {
    const float4* src = (const float4*)(kbase + (size_t)h * SINK * DHEAD);
    float4* dst = (float4*)(smem + KB_OFF);
    dst[t]       = src[t];
    dst[t + 256] = src[t + 256];
  }

  const unsigned short* hRow  = hb + (size_t)s0 * DIN;
  const unsigned short* wHead = wc + (size_t)h * COLS * DIN;

  // async-stage one K-chunk (A: 16x64B, B: 640x64B) into the given buffers
  auto stage = [&](int kc, unsigned aoff, unsigned boff) {
    if (t < 64) {
      int row = t >> 2;            // 0..15
      int c4  = t & 3;             // 16B segment within row
      async_g2l_b128(aoff + row * 64 + c4 * 16,
                     (unsigned long long)(hRow + (size_t)row * DIN + kc + c4 * 8));
    }
#pragma unroll
    for (int i = 0; i < 10; ++i) {
      int l   = t + 256 * i;
      int row = l >> 2;
      int c4  = l & 3;
      async_g2l_b128(boff + row * 64 + c4 * 16,
                     (unsigned long long)(wHead + (size_t)row * DIN + kc + c4 * 8));
    }
  };

  v8f acc[TPW] = {};

  stage(0, A0_OFF, B0_OFF);   // prologue: chunk 0 -> buffer 0

  for (int kc = 0; kc < DIN; kc += KC) {
    const int cur = (kc >> 5) & 1;
    const unsigned aoff = cur ? A1_OFF : A0_OFF;
    const unsigned boff = cur ? B1_OFF : B0_OFF;

    wait_asynccnt0();   // this wave's chunk-kc DMA landed in LDS
    __syncthreads();    // all waves landed; all waves done reading buf^1

    if (kc + KC < DIN)  // overlap next chunk's DMA with this chunk's WMMAs
      stage(kc + KC, cur ? A0_OFF : A1_OFF, cur ? B0_OFF : B1_OFF);

    // A fragment per ISA 16-bit A 16x32 layout:
    // lanes<16: K = {0..7,16..23}; lanes>=16: K = {8..15,24..31}.
    union Frag { v16bf v; uint4 q[2]; };
    Frag af;
    {
      int m  = lane & 15;
      int hi = lane >> 4;
      const unsigned char* p = smem + aoff + m * 64 + hi * 16;
      af.q[0] = *(const uint4*)(p);
      af.q[1] = *(const uint4*)(p + 32);
    }
    // load ALL B fragments, then fence the scheduler, then WMMA burst
    Frag bfm[TPW];
#pragma unroll
    for (int j = 0; j < TPW; ++j) {
      int tn = wave + NWAVE * j;
      int n  = tn * 16 + (lane & 15);
      const unsigned char* p = smem + boff + n * 64 + (lane >> 4) * 16;
      bfm[j].q[0] = *(const uint4*)(p);
      bfm[j].q[1] = *(const uint4*)(p + 32);
    }
    __builtin_amdgcn_sched_barrier(0);   // keep loads above, WMMAs below
#pragma unroll
    for (int j = 0; j < TPW; ++j)
      acc[j] = __builtin_amdgcn_wmma_f32_16x16x32_bf16(
          false, af.v, false, bfm[j].v, (short)0, acc[j], false, false);
  }

  // ---- stage accumulators to LDS f32 [16][QK_STRIDE] (reuses A/B0 region) -
  __syncthreads();
  float* qk = (float*)(smem);
#pragma unroll
  for (int j = 0; j < TPW; ++j) {
    int tn = wave + NWAVE * j;
    int n  = tn * 16 + (lane & 15);
    int mb = (lane >> 4) * 8;      // C layout: VGPR r -> M=r (lanes<16), r+8 (lanes>=16)
#pragma unroll
    for (int r = 0; r < 8; ++r)
      qk[(mb + r) * QK_STRIDE + n] = acc[j][r];
  }
  __syncthreads();

  // add q bias (bq) in place: 16*512 elements, 32 per thread
#pragma unroll
  for (int i = 0; i < 32; ++i) {
    int l = t + 256 * i;
    int s = l >> 9;
    int c = l & 511;
    qk[s * QK_STRIDE + c] += bq[h * QCOLS + c];
  }
  __syncthreads();

  // ---- epilogue: RMSNorm, logits, sink gate (64 threads: s x g) -----------
  if (t < 64) {
    const int s = t & 15;
    const int g = t >> 4;
    const float* qrow = qk + s * QK_STRIDE + g * DHEAD;
    const float* krow = qk + s * QK_STRIDE + QCOLS;
    const float* kbl  = (const float*)(smem + KB_OFF);

    float ssq_q = 0.f, ssq_k = 0.f;
    for (int d = 0; d < DHEAD; ++d) {
      float a = qrow[d], c = krow[d];
      ssq_q += a * a;
      ssq_k += c * c;
    }
    const float rq = rsqrtf(ssq_q * (1.f / DHEAD) + 1e-6f);
    const float rk = rsqrtf(ssq_k * (1.f / DHEAD) + 1e-6f);
    const float inv_d = 0.088388347648318447f;   // 1/sqrt(128)

    float dot = 0.f;
    float lb[SINK];
#pragma unroll
    for (int u = 0; u < SINK; ++u) lb[u] = 0.f;
    for (int d = 0; d < DHEAD; ++d) {
      float qn = qrow[d] * rq * qw[d];
      dot += qn * (krow[d] * rk * kw[d]);
#pragma unroll
      for (int u = 0; u < SINK; ++u) lb[u] += kbl[u * DHEAD + d] * qn;
    }
    const float logit = dot * inv_d + 2.f * bb[h * NGROUP + g];
    float se = 0.f;
#pragma unroll
    for (int u = 0; u < SINK; ++u) se += __expf(lb[u] * inv_d - logit);
    ((float*)(smem + SC_OFF))[s * 4 + g] = 1.f / (1.f + se);
  }
  __syncthreads();
  if (t < 16) {
    const float* sc = (const float*)(smem + SC_OFF);
    out[(size_t)h * NSEQ + s0 + t] =
        0.25f * (sc[t * 4] + sc[t * 4 + 1] + sc[t * 4 + 2] + sc[t * 4 + 3]);
  }
}

// ---------------------------------------------------------------------------
extern "C" void kernel_launch(void* const* d_in, const int* in_sizes, int n_in,
                              void* d_out, int out_size, void* d_ws, size_t ws_size,
                              hipStream_t stream) {
  const float* hidden = (const float*)d_in[0];
  const float* Wq     = (const float*)d_in[1];
  const float* bq     = (const float*)d_in[2];
  const float* Wk     = (const float*)d_in[3];
  const float* qw     = (const float*)d_in[4];
  const float* kw     = (const float*)d_in[5];
  const float* bb     = (const float*)d_in[6];
  const float* kbase  = (const float*)d_in[7];

  unsigned short* hb = (unsigned short*)d_ws;                         // 134 MB
  unsigned short* wc = (unsigned short*)((char*)d_ws +
                        (size_t)NSEQ * DIN * sizeof(unsigned short)); // +21 MB

  // hidden: 67,108,864 elems / 4 per thread / 256 = 65536 blocks
  cvt_hidden_bf16<<<65536, 256, 0, stream>>>(hidden, hb);
  // weights: 10,485,760 elems / 4 per thread / 256 = 10240 blocks
  pack_weights_bf16<<<10240, 256, 0, stream>>>(Wq, Wk, wc);

  dim3 grid(NSEQ / 16, NHEAD);
  fused_qk_gate<<<grid, 256, 0, stream>>>(hb, wc, bq, qw, kw, bb, kbase,
                                          (float*)d_out);
}
